// MultiheadAttention_712964571682
// MI455X (gfx1250) — compile-verified, matrix operand order checked by reading, not run on hardware
//
#include <hip/hip_runtime.h>

// ---------------------------------------------------------------------------
// MI455X (gfx1250) MHA forward: bf16 WMMA GEMMs + flash attention, with
// CDNA5 async global->LDS tile staging (ASYNCcnt path).
// B=1, S=2048, D=1024, H=16, HD=64, MAX_DIST=1024.
// ---------------------------------------------------------------------------

typedef __attribute__((ext_vector_type(16))) __bf16 v16bf;
typedef __attribute__((ext_vector_type(8)))  float  v8f;

union Frag { v16bf v; uint4 u[2]; };

__device__ __forceinline__ unsigned short f2bf(float f) {
    unsigned int u = __float_as_uint(f);
    u += 0x7FFFu + ((u >> 16) & 1u);            // round-to-nearest-even
    return (unsigned short)(u >> 16);
}

// CDNA5 async copy: 16B global -> LDS, tracked by ASYNCcnt (no VGPR data).
__device__ __forceinline__ void async_ld_b128(const void* smem, const void* gmem) {
    unsigned loff = (unsigned)(unsigned long long)(uintptr_t)smem;
    unsigned long long gaddr = (unsigned long long)(uintptr_t)gmem;
    asm volatile("global_load_async_to_lds_b128 %0, %1, off"
                 :: "v"(loff), "v"(gaddr) : "memory");
}
__device__ __forceinline__ void wait_async() {
    asm volatile("s_wait_asynccnt 0x0" ::: "memory");
}

#define S_LEN 2048
#define DMODEL 1024
#define NHEAD 16
#define HD 64

// ---------------------------------------------------------------------------
// Kernel 1: M = G_total @ rotation_matrix (64x64); thread t owns row t.
// ---------------------------------------------------------------------------
__global__ void prep_M_kernel(const float* __restrict__ thetas,
                              const float* __restrict__ theta_scale,
                              const int*   __restrict__ rot_idx,
                              const float* __restrict__ R,
                              float* __restrict__ Mout) {
    __shared__ float As[64 * 64];
    __shared__ float Rs[64 * 64];
    int t = threadIdx.x; // 64 threads
    for (int i = t; i < 64 * 64; i += 64) Rs[i] = R[i];
    for (int c = 0; c < 64; ++c) As[t * 64 + c] = (c == t) ? 1.f : 0.f;
    float ts = theta_scale[0];
    for (int r = 0; r < 32; ++r) {
        float th = thetas[r] * ts;
        float cth = cosf(th), sth = sinf(th);
        int i = rot_idx[2 * r], j = rot_idx[2 * r + 1];
        if (i == j) {
            As[t * 64 + i] *= cth;   // sequential .at updates collapse
        } else {
            float ai = As[t * 64 + i], aj = As[t * 64 + j];
            As[t * 64 + i] =  cth * ai + sth * aj;
            As[t * 64 + j] = -sth * ai + cth * aj;
        }
    }
    __syncthreads();
    for (int c = 0; c < 64; ++c) {
        float acc = 0.f;
        for (int d = 0; d < 64; ++d) acc += As[t * 64 + d] * Rs[d * 64 + c];
        Mout[t * 64 + c] = acc;
    }
}

// ---------------------------------------------------------------------------
// Kernel 2: Wrot^T[h*64+c][row] = sum_d W[row][h*64+d] * M[d][c]  (bf16).
// Output is stored N-major so GEMM B-tiles are contiguous strips.
// ---------------------------------------------------------------------------
__global__ void fold_W_kernel(const float* __restrict__ W,
                              const float* __restrict__ M,
                              unsigned short* __restrict__ WrotT) {
    __shared__ float Ms[64 * 64];
    int t = threadIdx.x;
    for (int i = t; i < 4096; i += 256) Ms[i] = M[i];
    __syncthreads();
    int h = blockIdx.y;
    int rl = t >> 6, c = t & 63;
    int row = blockIdx.x * 4 + rl;
    const float* wrow = W + (size_t)row * DMODEL + h * HD;
    float acc = 0.f;
    #pragma unroll 8
    for (int d = 0; d < 64; ++d) acc += wrow[d] * Ms[d * 64 + c];
    WrotT[(size_t)(h * HD + c) * DMODEL + row] = f2bf(acc);
}

__global__ void fold_bias_kernel(const float* __restrict__ b,
                                 const float* __restrict__ M,
                                 float* __restrict__ brot) {
    int idx = blockIdx.x * blockDim.x + threadIdx.x; // 1024
    int h = idx >> 6, c = idx & 63;
    float acc = 0.f;
    for (int d = 0; d < 64; ++d) acc += b[h * HD + d] * M[d * 64 + c];
    brot[idx] = acc;
}

__global__ void conv_bf16_kernel(const float* __restrict__ src,
                                 unsigned short* __restrict__ dst, int n) {
    int i = blockIdx.x * blockDim.x + threadIdx.x;
    if (i < n) dst[i] = f2bf(src[i]);
}

// LDS-tiled fp32 -> bf16 transpose: dst[c][r] = src[r][c]  (R x C).
__global__ void convT_bf16_kernel(const float* __restrict__ src,
                                  unsigned short* __restrict__ dst,
                                  int R, int C) {
    __shared__ float tile[32][33];
    int r0 = blockIdx.y * 32, c0 = blockIdx.x * 32;
    int tr = threadIdx.x & 31, tc = threadIdx.x >> 5; // 256 thr: 8 rows/pass
    #pragma unroll
    for (int rr = tc; rr < 32; rr += 8)
        tile[rr][tr] = src[(size_t)(r0 + rr) * C + c0 + tr];
    __syncthreads();
    #pragma unroll
    for (int rr = tc; rr < 32; rr += 8)
        dst[(size_t)(c0 + rr) * R + r0 + tr] = f2bf(tile[tr][rr]);
}

// ---------------------------------------------------------------------------
// Generic bf16 WMMA GEMM: C = A(MxK) @ Bt^T  (Bt given N-major, NxK).
// Block 128 threads (4 wave32), tile 64x64, K-step 32.
// Tiles staged with global_load_async_to_lds_b128 (ASYNCcnt).
//   mode 0: fp32 out (+bias)          — final projection
//   mode 2: bf16 out, RoPE epilogue   — Q/K projections (N-tile == head)
//   mode 3: bf16 out, transposed      — V projection (produces vT)
// ---------------------------------------------------------------------------
__global__ __launch_bounds__(128) void gemm_bf16_kernel(
    const unsigned short* __restrict__ A, const unsigned short* __restrict__ Bt,
    const float* __restrict__ bias, void* __restrict__ Cout,
    int M, int N, int K, int mode,
    const float* __restrict__ inv_freq, const float* __restrict__ pscale) {

    __shared__ __align__(16) union {
        struct { unsigned short Al[64 * 32]; unsigned short Bl[64 * 32]; } ab;
        float Cs[64 * 64];
    } sm;

    int t = threadIdx.x;
    int wave = t >> 5, lane = t & 31;
    int m0 = blockIdx.x * 64, n0 = blockIdx.y * 64;
    int half = (lane < 16) ? 0 : 8;
    int lr = lane & 15;

    // per-thread tile slots (two 16B chunks of each 64x32 tile)
    int r0 = t >> 2,          s0 = (t & 3) * 8;
    int r1 = (t + 128) >> 2,  s1 = ((t + 128) & 3) * 8;

    v8f acc[4];
    #pragma unroll
    for (int i = 0; i < 4; ++i)
        #pragma unroll
        for (int j = 0; j < 8; ++j) acc[i][j] = 0.f;

    for (int k0 = 0; k0 < K; k0 += 32) {
        __syncthreads();
        async_ld_b128(&sm.ab.Al[r0 * 32 + s0], &A[(size_t)(m0 + r0) * K + k0 + s0]);
        async_ld_b128(&sm.ab.Al[r1 * 32 + s1], &A[(size_t)(m0 + r1) * K + k0 + s1]);
        async_ld_b128(&sm.ab.Bl[r0 * 32 + s0], &Bt[(size_t)(n0 + r0) * K + k0 + s0]);
        async_ld_b128(&sm.ab.Bl[r1 * 32 + s1], &Bt[(size_t)(n0 + r1) * K + k0 + s1]);
        wait_async();
        __syncthreads();

        int mrow = wave * 16 + lr;
        Frag fa;
        fa.u[0] = *(const uint4*)&sm.ab.Al[mrow * 32 + half];
        fa.u[1] = *(const uint4*)&sm.ab.Al[mrow * 32 + half + 16];
        #pragma unroll
        for (int nt = 0; nt < 4; ++nt) {
            int ncol = nt * 16 + lr;
            Frag fb;
            fb.u[0] = *(const uint4*)&sm.ab.Bl[ncol * 32 + half];
            fb.u[1] = *(const uint4*)&sm.ab.Bl[ncol * 32 + half + 16];
            acc[nt] = __builtin_amdgcn_wmma_f32_16x16x32_bf16(
                false, fa.v, false, fb.v, (short)0, acc[nt], false, false);
        }
    }
    __syncthreads();
    // stage C to LDS (C layout: lane<16 -> M=r, lane>=16 -> M=8+r, N=lane%16)
    {
        int mbase = wave * 16 + half;
        #pragma unroll
        for (int nt = 0; nt < 4; ++nt) {
            int ncol = nt * 16 + lr;
            #pragma unroll
            for (int r = 0; r < 8; ++r)
                sm.Cs[(mbase + r) * 64 + ncol] = acc[nt][r];
        }
    }
    __syncthreads();

    if (mode == 2) {
        float ps = pscale[0];
        #pragma unroll
        for (int it = 0; it < 32; ++it) {
            int i = t + it * 128;
            int row = i >> 6, c = i & 63;
            int f = c & 31;
            float b1 = bias ? bias[n0 + 2 * f] : 0.f;
            float b2 = bias ? bias[n0 + 2 * f + 1] : 0.f;
            float x1 = sm.Cs[row * 64 + 2 * f] + b1;
            float x2 = sm.Cs[row * 64 + 2 * f + 1] + b2;
            float ang = (float)(m0 + row) * inv_freq[f];
            float cs = cosf(ang), sn = sinf(ang);
            float val = (c < 32) ? (x1 * cs - x2 * sn) : (x1 * sn + x2 * cs);
            ((unsigned short*)Cout)[(size_t)(m0 + row) * N + n0 + c] = f2bf(val * ps);
        }
    } else if (mode == 3) {
        #pragma unroll
        for (int it = 0; it < 32; ++it) {
            int i = t + it * 128;
            int row = i >> 6, c = i & 63;
            float val = sm.Cs[i] + (bias ? bias[n0 + c] : 0.f);
            ((unsigned short*)Cout)[(size_t)(n0 + c) * M + m0 + row] = f2bf(val);
        }
    } else {
        #pragma unroll
        for (int it = 0; it < 32; ++it) {
            int i = t + it * 128;
            int row = i >> 6, c = i & 63;
            float val = sm.Cs[i] + (bias ? bias[n0 + c] : 0.f);
            ((float*)Cout)[(size_t)(m0 + row) * N + n0 + c] = val;
        }
    }
}

// ---------------------------------------------------------------------------
// Flash attention per (64-query block, head). 128 threads = 4 waves, each
// wave owns 16 query rows; key blocks of 32. V supplied transposed
// (vT[1024][2048]) so every tile stages via async b128 loads.
// ---------------------------------------------------------------------------
__global__ __launch_bounds__(128) void attn_kernel(
    const unsigned short* __restrict__ qb, const unsigned short* __restrict__ kb,
    const unsigned short* __restrict__ vT, const float* __restrict__ relb,
    unsigned short* __restrict__ ab, float* __restrict__ qk_out) {

    __shared__ __align__(16) unsigned short Qs[64 * 64];    // [qrow][d]
    __shared__ __align__(16) unsigned short Ks[32 * 64];    // [key][d]
    __shared__ __align__(16) unsigned short Vt[64 * 32];    // [d(col)][key]
    __shared__ __align__(16) unsigned short Ps[4][16 * 32]; // per-wave P tile

    int t = threadIdx.x, wave = t >> 5, lane = t & 31;
    int half = (lane < 16) ? 0 : 8;
    int lr = lane & 15;
    int Moff = (lane < 16) ? 0 : 8;
    int q0 = blockIdx.x * 64;
    int h  = blockIdx.y;

    // Q tile: 64 rows x 128B -> 512 x 16B chunks, 4 per thread (async)
    #pragma unroll
    for (int u = 0; u < 4; ++u) {
        int i = t + u * 128;
        int row = i >> 3, seg = (i & 7) * 8;
        async_ld_b128(&Qs[row * 64 + seg],
                      &qb[(size_t)(q0 + row) * DMODEL + h * HD + seg]);
    }
    wait_async();
    __syncthreads();

    Frag aq0, aq1;  // A-frags for d chunks 0-31, 32-63
    {
        int mrow = wave * 16 + lr;
        aq0.u[0] = *(const uint4*)&Qs[mrow * 64 + half];
        aq0.u[1] = *(const uint4*)&Qs[mrow * 64 + half + 16];
        aq1.u[0] = *(const uint4*)&Qs[mrow * 64 + 32 + half];
        aq1.u[1] = *(const uint4*)&Qs[mrow * 64 + 32 + half + 16];
    }

    v8f O[4];
    #pragma unroll
    for (int i = 0; i < 4; ++i)
        #pragma unroll
        for (int j = 0; j < 8; ++j) O[i][j] = 0.f;
    float mx[8], ls[8];
    #pragma unroll
    for (int r = 0; r < 8; ++r) { mx[r] = -3.0e38f; ls[r] = 0.f; }

    const float S1 = 0.35355339059327373f;   // hd^-0.25
    const float S2 = 0.47855339059327373f;   // hd^-0.5 + hd^-0.25

    // per-thread K/V tile slots
    int kr0 = t >> 2,          ks0 = (t & 3) * 8;          // K: 32x64 -> 256 chunks? (128B rows)
    int kr1 = -1, ks1 = -1;
    {   // K tile: 32 rows x 128B = 256 chunks -> 2 per thread
        int i1 = t + 128;
        kr1 = i1 >> 3; ks1 = (i1 & 7) * 8;
        kr0 = t >> 3;  ks0 = (t & 7) * 8;
    }
    int vr0 = t >> 2,          vs0 = (t & 3) * 8;          // Vt: 64 rows x 64B = 256 chunks
    int vr1 = (t + 128) >> 2,  vs1 = ((t + 128) & 3) * 8;

    for (int kblk = 0; kblk < S_LEN / 32; ++kblk) {
        __syncthreads();
        size_t kbase = (size_t)(kblk * 32) * DMODEL + h * HD;
        async_ld_b128(&Ks[kr0 * 64 + ks0], &kb[kbase + (size_t)kr0 * DMODEL + ks0]);
        async_ld_b128(&Ks[kr1 * 64 + ks1], &kb[kbase + (size_t)kr1 * DMODEL + ks1]);
        size_t vbase = (size_t)(h * HD) * S_LEN + kblk * 32;
        async_ld_b128(&Vt[vr0 * 32 + vs0], &vT[vbase + (size_t)vr0 * S_LEN + vs0]);
        async_ld_b128(&Vt[vr1 * 32 + vs1], &vT[vbase + (size_t)vr1 * S_LEN + vs1]);
        wait_async();
        __syncthreads();

        // raw scores: 16 queries x 32 keys (two 16x16 tiles)
        v8f sc[2];
        #pragma unroll
        for (int nt = 0; nt < 2; ++nt) {
            #pragma unroll
            for (int j = 0; j < 8; ++j) sc[nt][j] = 0.f;
            int key = nt * 16 + lr;
            Frag fb0, fb1;   // B[d][key] = K[key][d]
            fb0.u[0] = *(const uint4*)&Ks[key * 64 + half];
            fb0.u[1] = *(const uint4*)&Ks[key * 64 + half + 16];
            fb1.u[0] = *(const uint4*)&Ks[key * 64 + 32 + half];
            fb1.u[1] = *(const uint4*)&Ks[key * 64 + 32 + half + 16];
            sc[nt] = __builtin_amdgcn_wmma_f32_16x16x32_bf16(
                false, aq0.v, false, fb0.v, (short)0, sc[nt], false, false);
            sc[nt] = __builtin_amdgcn_wmma_f32_16x16x32_bf16(
                false, aq1.v, false, fb1.v, (short)0, sc[nt], false, false);
        }

        // bias + scales; emit qk rows (q%16==0 -> r==0 && lane<16)
        float lg[2][8];
        #pragma unroll
        for (int nt = 0; nt < 2; ++nt) {
            int key = kblk * 32 + nt * 16 + lr;
            #pragma unroll
            for (int r = 0; r < 8; ++r) {
                int qrow = q0 + wave * 16 + Moff + r;
                int dpos = qrow - key;
                dpos = (dpos > 1023) ? 1023 : ((dpos < -1023) ? -1023 : dpos);
                float bias = relb[(dpos + 1023) * NHEAD + h];
                float raw = sc[nt][r];
                lg[nt][r] = raw * S2 + bias;
                if (r == 0 && lane < 16 && key < 1024) {
                    int srow = h * 128 + (qrow >> 4);
                    qk_out[(size_t)srow * 1024 + key] = raw * S1 + bias;
                }
            }
        }

        // online softmax (row reductions across the 16-lane groups)
        float pw[2][8];
        #pragma unroll
        for (int r = 0; r < 8; ++r) {
            float vmax = fmaxf(lg[0][r], lg[1][r]);
            for (int s = 1; s < 16; s <<= 1)
                vmax = fmaxf(vmax, __shfl_xor(vmax, s, 16));
            float mnew = fmaxf(mx[r], vmax);
            float corr = __expf(mx[r] - mnew);
            mx[r] = mnew;
            float p0 = __expf(lg[0][r] - mnew);
            float p1 = __expf(lg[1][r] - mnew);
            pw[0][r] = p0; pw[1][r] = p1;
            float psum = p0 + p1;
            for (int s = 1; s < 16; s <<= 1)
                psum += __shfl_xor(psum, s, 16);
            ls[r] = ls[r] * corr + psum;
            #pragma unroll
            for (int nt = 0; nt < 4; ++nt) O[nt][r] *= corr;
        }

        // stage P (bf16) through LDS to re-enter A-fragment layout
        #pragma unroll
        for (int nt = 0; nt < 2; ++nt)
            #pragma unroll
            for (int r = 0; r < 8; ++r)
                Ps[wave][(Moff + r) * 32 + nt * 16 + lr] = f2bf(pw[nt][r]);
        __syncthreads();

        // O += P (16x32) @ V (32x64)
        Frag fp;
        fp.u[0] = *(const uint4*)&Ps[wave][lr * 32 + half];
        fp.u[1] = *(const uint4*)&Ps[wave][lr * 32 + half + 16];
        #pragma unroll
        for (int nt = 0; nt < 4; ++nt) {
            Frag fv;  // B[key][col] read from Vt[col][key]
            fv.u[0] = *(const uint4*)&Vt[(nt * 16 + lr) * 32 + half];
            fv.u[1] = *(const uint4*)&Vt[(nt * 16 + lr) * 32 + half + 16];
            O[nt] = __builtin_amdgcn_wmma_f32_16x16x32_bf16(
                false, fp.v, false, fv.v, (short)0, O[nt], false, false);
        }
    }

    // normalize, store attention output as bf16 [s][h*64+col]
    float inv[8];
    #pragma unroll
    for (int r = 0; r < 8; ++r) inv[r] = 1.f / ls[r];
    #pragma unroll
    for (int nt = 0; nt < 4; ++nt)
        #pragma unroll
        for (int r = 0; r < 8; ++r) {
            int qrow = q0 + wave * 16 + Moff + r;
            int col = h * HD + nt * 16 + lr;
            ab[(size_t)qrow * DMODEL + col] = f2bf(O[nt][r] * inv[r]);
        }
}

// ---------------------------------------------------------------------------
// Orchestration
// ---------------------------------------------------------------------------
extern "C" void kernel_launch(void* const* d_in, const int* in_sizes, int n_in,
                              void* d_out, int out_size, void* d_ws, size_t ws_size,
                              hipStream_t stream) {
    (void)in_sizes; (void)n_in; (void)out_size; (void)ws_size;

    const float* x        = (const float*)d_in[0];
    const float* Wq       = (const float*)d_in[1];
    const float* bq       = (const float*)d_in[2];
    const float* Wk       = (const float*)d_in[3];
    const float* Wv       = (const float*)d_in[4];
    const float* bv       = (const float*)d_in[5];
    const float* Wo       = (const float*)d_in[6];
    const float* bo       = (const float*)d_in[7];
    const float* thetas   = (const float*)d_in[8];
    const float* tscale   = (const float*)d_in[9];
    const float* rotmat   = (const float*)d_in[10];
    const float* inv_freq = (const float*)d_in[11];
    const float* pscale   = (const float*)d_in[12];
    const float* relb     = (const float*)d_in[13];
    const int*   rot_idx  = (const int*)d_in[14];

    // workspace carve-up (all offsets 256B aligned) — ~29 MB total
    char* w = (char*)d_ws;
    float* Mbuf  = (float*)(w);                         // 16384
    float* bqrot = (float*)(w + 16384);                 // 4096
    unsigned short* xb   = (unsigned short*)(w + 20480);            // 4 MB
    unsigned short* wqrT = (unsigned short*)(w + 20480 + 4194304);  // 2 MB (N-major)
    unsigned short* wkrT = (unsigned short*)(w + 20480 + 6291456);  // 2 MB (N-major)
    unsigned short* wvT  = (unsigned short*)(w + 20480 + 8388608);  // 2 MB (N-major)
    unsigned short* woT  = (unsigned short*)(w + 20480 + 10485760); // 2 MB (N-major)
    unsigned short* qbuf = (unsigned short*)(w + 20480 + 12582912); // 4 MB
    unsigned short* kbuf = (unsigned short*)(w + 20480 + 16777216); // 4 MB
    unsigned short* vTb  = (unsigned short*)(w + 20480 + 20971520); // 4 MB (vT[1024][2048])
    unsigned short* abuf = (unsigned short*)(w + 20480 + 25165824); // 4 MB

    float* out = (float*)d_out;                 // (1,2048,1024) fp32
    float* qk  = out + (size_t)S_LEN * DMODEL;  // (1,2048,1024) fp32

    // 1) combined rotation matrix M = G_total @ R
    prep_M_kernel<<<1, 64, 0, stream>>>(thetas, tscale, rot_idx, rotmat, Mbuf);

    // 2) fold M into Wq/Wk (bf16, transposed); fold bias; bf16 casts
    fold_W_kernel<<<dim3(256, 16), 256, 0, stream>>>(Wq, Mbuf, wqrT);
    fold_W_kernel<<<dim3(256, 16), 256, 0, stream>>>(Wk, Mbuf, wkrT);
    fold_bias_kernel<<<4, 256, 0, stream>>>(bq, Mbuf, bqrot);
    conv_bf16_kernel<<<(S_LEN * DMODEL) / 256, 256, 0, stream>>>(x, xb, S_LEN * DMODEL);
    convT_bf16_kernel<<<dim3(32, 32), 256, 0, stream>>>(Wv, wvT, DMODEL, DMODEL);
    convT_bf16_kernel<<<dim3(32, 32), 256, 0, stream>>>(Wo, woT, DMODEL, DMODEL);

    // 3) QKV projections (WMMA GEMM); Q/K with RoPE + positional scaling;
    //    V written transposed for async attention staging.
    dim3 ggrid(S_LEN / 64, DMODEL / 64);
    gemm_bf16_kernel<<<ggrid, 128, 0, stream>>>(xb, wqrT, bqrot, qbuf,
        S_LEN, DMODEL, DMODEL, 2, inv_freq, pscale);
    gemm_bf16_kernel<<<ggrid, 128, 0, stream>>>(xb, wkrT, nullptr, kbuf,
        S_LEN, DMODEL, DMODEL, 2, inv_freq, pscale);
    gemm_bf16_kernel<<<ggrid, 128, 0, stream>>>(xb, wvT, bv, vTb,
        S_LEN, DMODEL, DMODEL, 3, nullptr, nullptr);

    // 4) flash attention + qk side output
    attn_kernel<<<dim3(S_LEN / 64, NHEAD), 128, 0, stream>>>(
        qbuf, kbuf, vTb, relb, abuf, qk);

    // 5) output projection (fp32 out + bias)
    gemm_bf16_kernel<<<ggrid, 128, 0, stream>>>(abuf, woT, bo, out,
        S_LEN, DMODEL, DMODEL, 0, nullptr, nullptr);
}
